// MultiHeadGroupedQueryAttention_35399120453770
// MI455X (gfx1250) — compile-verified, matrix operand order checked by reading, not run on hardware
//
#include <hip/hip_runtime.h>
#include <hip/hip_bf16.h>

// ---------------------------------------------------------------------------
// MI455X (gfx1250) GQA attention: bf16 WMMA pipeline + TDM (tensor_load_to_lds)
// double-buffered tile movement.
// ---------------------------------------------------------------------------

typedef __attribute__((ext_vector_type(16))) __bf16 v16bf;
typedef __attribute__((ext_vector_type(8)))  __bf16 v8bf;
typedef __attribute__((ext_vector_type(8)))  float  v8f;
typedef unsigned int u32x4 __attribute__((ext_vector_type(4)));
typedef int          i32x8 __attribute__((ext_vector_type(8)));
typedef int          i32x4 __attribute__((ext_vector_type(4)));

#define DEVI __device__ __forceinline__

constexpr int Bc   = 2;
constexpr int Nc   = 2048;
constexpr int Ec   = 1024;
constexpr int HQc  = 16;
constexpr int HKVc = 4;
constexpr int Gc   = 4;
constexpr int KVEc = 256;   // HKV * D ; D = 64, scale = 1/sqrt(64)

#if __has_builtin(__builtin_amdgcn_tensor_load_to_lds) && \
    __has_builtin(__builtin_amdgcn_s_wait_tensorcnt)
#define HAVE_TDM 1
#else
#define HAVE_TDM 0
#endif

DEVI unsigned short f2bf(float f) {
    unsigned int u = __float_as_uint(f);
    u += 0x7FFFu + ((u >> 16) & 1u);          // round-to-nearest-even
    return (unsigned short)(u >> 16);
}

DEVI v8f wmma_bf(v16bf a, v16bf b, v8f c) {
    return __builtin_amdgcn_wmma_f32_16x16x32_bf16(
        false, a, false, b, (short)0, c, false, false);
}

// WMMA 16-bit A/B operand (16x32) from a row-major LDS tile: two contiguous
// 16B runs per lane (lane<16: K 0..7 & 16..23 ; lane>=16: K 8..15 & 24..31).
DEVI v16bf load_ab(const __bf16* base, int lane, int k0, int ldw) {
    int row = lane & 15;
    int kh  = (lane >> 4) << 3;
    const __bf16* p = base + row * ldw + k0 + kh;
    union { v16bf v; v8bf h[2]; } u;
    u.h[0] = *(const v8bf*)(p);
    u.h[1] = *(const v8bf*)(p + 16);
    return u.v;
}

DEVI float rmax16(float v) {
#pragma unroll
    for (int m = 1; m < 16; m <<= 1) v = fmaxf(v, __shfl_xor(v, m, 32));
    return v;
}
DEVI float rsum16(float v) {
#pragma unroll
    for (int m = 1; m < 16; m <<= 1) v += __shfl_xor(v, m, 32);
    return v;
}

// ---------------------------------------------------------------------------
// Tile movement: TDM DMA (global -> LDS, 2-D strided tile) when available,
// else cooperative vector copy. tileW elems per row, tileH rows, row stride
// strideE elems, bf16 data.
// ---------------------------------------------------------------------------
DEVI void tile_load(__bf16* dstLds, const __bf16* srcG,
                    int tileW, int tileH, int strideE, int tid, int wave) {
#if HAVE_TDM
    if (wave == 0) {
        unsigned long long ga = (unsigned long long)(uintptr_t)srcG;
        unsigned lds = (unsigned)(uintptr_t)dstLds;     // low 32 bits = LDS addr
        u32x4 g0 = { 1u,                                 // count=1 (valid D#)
                     lds,
                     (unsigned)(ga & 0xFFFFFFFFu),
                     (unsigned)((ga >> 32) & 0x01FFFFFFu) | (2u << 30) }; // type=2
        unsigned tw = (unsigned)tileW, th = (unsigned)tileH;
        i32x8 g1 = { (int)0x00010000u,                   // data_size=1 (2 bytes)
                     (int)((tw & 0xFFFFu) << 16),        // tensor_dim0 lo
                     (int)((tw >> 16) | ((th & 0xFFFFu) << 16)),  // d0 hi | d1 lo
                     (int)((th >> 16) | ((tw & 0xFFFFu) << 16)),  // d1 hi | tile_dim0
                     (int)(th & 0xFFFFu),                // tile_dim1 (tile_dim2=0)
                     (int)(unsigned)strideE,             // tensor_dim0_stride lo32
                     0, 0 };
        i32x4 gz4 = { 0, 0, 0, 0 };
        i32x8 gz8 = { 0, 0, 0, 0, 0, 0, 0, 0 };
        __builtin_amdgcn_tensor_load_to_lds(g0, g1, gz4, gz4, gz8, 0);
    }
#else
    int chunks = (tileW * tileH) >> 3;
    int cw = tileW >> 3;                                 // v8bf chunks per row
    for (int c = tid; c < chunks; c += 128) {
        int row = c / cw, col8 = (c % cw) << 3;
        *(v8bf*)&dstLds[row * tileW + col8] =
            *(const v8bf*)&srcG[(size_t)row * strideE + col8];
    }
#endif
}

DEVI void tile_wait(int wave) {
#if HAVE_TDM
    if (wave == 0) __builtin_amdgcn_s_wait_tensorcnt(0);
#else
    (void)wave;
#endif
}

// ---------------------------------------------------------------------------
__global__ void k_cvt_bf16(const float* __restrict__ in,
                           unsigned short* __restrict__ out, int n) {
    int i = blockIdx.x * blockDim.x + threadIdx.x;
    if (i < n) out[i] = f2bf(in[i]);
}

__global__ void k_zero(float* __restrict__ p, int n) {
    int i = blockIdx.x * blockDim.x + threadIdx.x;
    if (i < n) p[i] = 0.f;
}

// ---------------------------------------------------------------------------
// GEMM: C[M x Nout] = A[M x K](bf16) * W[K x Nout](bf16). 64x64 tile, 4 waves.
// A tiles: TDM double-buffered. W tiles: transposed into LDS (VGPR path),
// double-buffered, issued in the prefetch slot.
// ---------------------------------------------------------------------------
template <bool OUT_F32>
__global__ void __launch_bounds__(128)
k_gemm(const __bf16* __restrict__ A, const __bf16* __restrict__ W,
       void* __restrict__ Cout, int M, int K, int Nout) {
    __shared__ __align__(16) __bf16 As [2][64 * 32];
    __shared__ __align__(16) __bf16 Wts[2][64 * 32];

    int tid  = threadIdx.x;
    int lane = tid & 31, wave = tid >> 5;
    int br = blockIdx.y, bc = blockIdx.x;

    auto loadW = [&](int kb, int buf) {
        for (int c = tid; c < 256; c += 128) {
            int kr = c >> 3, n8 = (c & 7) << 3;
            v8bf wv = *(const v8bf*)&W[(size_t)(kb * 32 + kr) * Nout + bc * 64 + n8];
#pragma unroll
            for (int e = 0; e < 8; ++e) Wts[buf][(n8 + e) * 32 + kr] = wv[e];
        }
    };

    int nK = K >> 5;
    tile_load(As[0], &A[(size_t)(br * 64) * K], 32, 64, K, tid, wave);
    loadW(0, 0);

    v8f acc[4] = {};
    for (int kb = 0; kb < nK; ++kb) {
        int buf = kb & 1;
        tile_wait(wave);
        __syncthreads();
        if (kb + 1 < nK) {
            tile_load(As[buf ^ 1], &A[(size_t)(br * 64) * K + (kb + 1) * 32],
                      32, 64, K, tid, wave);
            loadW(kb + 1, buf ^ 1);
        }
        v16bf a = load_ab(&As[buf][wave * 16 * 32], lane, 0, 32);
#pragma unroll
        for (int nt = 0; nt < 4; ++nt) {
            v16bf bm = load_ab(&Wts[buf][nt * 16 * 32], lane, 0, 32);
            acc[nt] = wmma_bf(a, bm, acc[nt]);
        }
    }
    int kh = (lane >> 4) << 3;
#pragma unroll
    for (int nt = 0; nt < 4; ++nt) {
#pragma unroll
        for (int r = 0; r < 8; ++r) {
            int row = br * 64 + wave * 16 + r + kh;
            int col = bc * 64 + nt * 16 + (lane & 15);
            float v = acc[nt][r];
            if (OUT_F32)
                ((float*)Cout)[(size_t)row * Nout + col] = v;
            else
                ((unsigned short*)Cout)[(size_t)row * Nout + col] = f2bf(v);
        }
    }
}

// ---------------------------------------------------------------------------
// Attention: workgroup = (b, i=hkv, j=group, 64-row query block), 4 waves.
// Phase 1: exact row max/sumexp. Phase 2: normalized P, P*V, column sums
// (attn_w needs true column sums of normalized attention).
// q head = i*G + j; k head = i; v head = j; out channel = (j*4+i)*64 + d.
// K tiles via TDM double buffer; V transposed via VGPR path, double-buffered.
// ---------------------------------------------------------------------------
__global__ void __launch_bounds__(128)
k_attn(const __bf16* __restrict__ Q, const __bf16* __restrict__ Kb,
       const __bf16* __restrict__ Vb, float* __restrict__ y,
       float* __restrict__ attn_sum) {
    __shared__ __align__(16) __bf16 Qs [64 * 64];
    __shared__ __align__(16) __bf16 Ks [2][64 * 64];
    __shared__ __align__(16) __bf16 Vts[2][64 * 64];  // transposed: [d][s]
    __shared__ __align__(16) __bf16 Ps [64 * 64];
    __shared__ float colsum[Nc];

    int tid = threadIdx.x, lane = tid & 31, wave = tid >> 5;
    int qb = blockIdx.x;
    int iy = blockIdx.y;                 // q-head = i*G + j
    int b  = blockIdx.z;
    int i  = iy >> 2, j = iy & 3;
    const float scale = 0.125f;
    int kh = (lane >> 4) << 3;
    int nCols = (qb + 1) * 64;

    const __bf16* Kg = &Kb[(size_t)(b * Nc) * KVEc + i * 64];
    const __bf16* Vg = &Vb[(size_t)(b * Nc) * KVEc + j * 64];

    auto loadV = [&](int t, int buf) {   // global V tile -> transposed LDS
        for (int c = tid; c < 512; c += 128) {
            int row = c >> 3, col8 = (c & 7) << 3;
            v8bf vv = *(const v8bf*)&Vg[(size_t)(t * 64 + row) * KVEc + col8];
#pragma unroll
            for (int e = 0; e < 8; ++e) Vts[buf][(col8 + e) * 64 + row] = vv[e];
        }
    };

    for (int s = tid; s < nCols; s += 128) colsum[s] = 0.f;

    // Q tile + first K tile via TDM
    tile_load(Qs, &Q[(size_t)(b * Nc + qb * 64) * Ec + iy * 64], 64, 64, Ec, tid, wave);
    tile_load(Ks[0], Kg, 64, 64, KVEc, tid, wave);

    float rm[8], rl[8];
#pragma unroll
    for (int r = 0; r < 8; ++r) { rm[r] = -1e30f; rl[r] = 0.f; }

    v16bf qa0, qa1;
    // ---------------- Phase 1: row statistics ----------------
    for (int t = 0; t <= qb; ++t) {
        int buf = t & 1;
        tile_wait(wave);
        __syncthreads();
        if (t == 0) {
            qa0 = load_ab(&Qs[wave * 16 * 64], lane, 0, 64);
            qa1 = load_ab(&Qs[wave * 16 * 64], lane, 32, 64);
        }
        if (t < qb)
            tile_load(Ks[buf ^ 1], Kg + (size_t)(t + 1) * 64 * KVEc, 64, 64, KVEc, tid, wave);

        v8f sacc[4];
#pragma unroll
        for (int nt = 0; nt < 4; ++nt) {
            v8f s = {};
            v16bf b0 = load_ab(&Ks[buf][nt * 16 * 64], lane, 0, 64);
            s = wmma_bf(qa0, b0, s);
            v16bf b1 = load_ab(&Ks[buf][nt * 16 * 64], lane, 32, 64);
            s = wmma_bf(qa1, b1, s);
            int col = t * 64 + nt * 16 + (lane & 15);
#pragma unroll
            for (int r = 0; r < 8; ++r) {
                int row = qb * 64 + wave * 16 + r + kh;
                float v = s[r] * scale;
                s[r] = (col > row) ? -1e30f : v;
            }
            sacc[nt] = s;
        }
#pragma unroll
        for (int r = 0; r < 8; ++r) {
            float tmax = fmaxf(fmaxf(sacc[0][r], sacc[1][r]),
                               fmaxf(sacc[2][r], sacc[3][r]));
            tmax = rmax16(tmax);
            float mnew = fmaxf(rm[r], tmax);
            float sum = __expf(sacc[0][r] - mnew) + __expf(sacc[1][r] - mnew) +
                        __expf(sacc[2][r] - mnew) + __expf(sacc[3][r] - mnew);
            sum = rsum16(sum);
            rl[r] = rl[r] * __expf(rm[r] - mnew) + sum;
            rm[r] = mnew;
        }
    }
    float rinv[8];
#pragma unroll
    for (int r = 0; r < 8; ++r) rinv[r] = 1.0f / rl[r];

    v8f o[4] = {};

    // ---------------- Phase 2: P, P*V, column sums ----------------
    __syncthreads();                       // all waves done reading phase-1 tiles
    tile_load(Ks[0], Kg, 64, 64, KVEc, tid, wave);
    loadV(0, 0);
    for (int t = 0; t <= qb; ++t) {
        int buf = t & 1;
        tile_wait(wave);
        __syncthreads();
        if (t < qb) {
            tile_load(Ks[buf ^ 1], Kg + (size_t)(t + 1) * 64 * KVEc, 64, 64, KVEc, tid, wave);
            loadV(t + 1, buf ^ 1);
        }
#pragma unroll
        for (int nt = 0; nt < 4; ++nt) {
            v8f s = {};
            v16bf b0 = load_ab(&Ks[buf][nt * 16 * 64], lane, 0, 64);
            s = wmma_bf(qa0, b0, s);
            v16bf b1 = load_ab(&Ks[buf][nt * 16 * 64], lane, 32, 64);
            s = wmma_bf(qa1, b1, s);
            int col = t * 64 + nt * 16 + (lane & 15);
            float csum = 0.f;
#pragma unroll
            for (int r = 0; r < 8; ++r) {
                int row = qb * 64 + wave * 16 + r + kh;
                float v = s[r] * scale;
                v = (col > row) ? 0.f : __expf(v - rm[r]) * rinv[r];
                csum += v;
                ((unsigned short*)Ps)[(wave * 16 + r + kh) * 64 + nt * 16 + (lane & 15)] =
                    f2bf(v);
            }
            csum += __shfl_xor(csum, 16, 32);       // combine row halves
            if (lane < 16) atomicAdd(&colsum[col], csum);
        }
        // own-wave P strip: DS ops are in-order per wave, no barrier needed
        v16bf pa0 = load_ab(&Ps[wave * 16 * 64], lane, 0, 64);
        v16bf pa1 = load_ab(&Ps[wave * 16 * 64], lane, 32, 64);
#pragma unroll
        for (int dt = 0; dt < 4; ++dt) {
            v16bf vb0 = load_ab(&Vts[buf][dt * 16 * 64], lane, 0, 64);
            o[dt] = wmma_bf(pa0, vb0, o[dt]);
            v16bf vb1 = load_ab(&Vts[buf][dt * 16 * 64], lane, 32, 64);
            o[dt] = wmma_bf(pa1, vb1, o[dt]);
        }
    }
    __syncthreads();

    // O -> pre-LN buffer, merged head order (j,i)
#pragma unroll
    for (int dt = 0; dt < 4; ++dt) {
#pragma unroll
        for (int r = 0; r < 8; ++r) {
            int row = qb * 64 + wave * 16 + r + kh;
            int cch = (j * 4 + i) * 64 + dt * 16 + (lane & 15);
            y[(size_t)(b * Nc + row) * Ec + cch] = o[dt][r];
        }
    }
    float* asum = &attn_sum[(((size_t)b * HKVc + i) * Gc + j) * Nc];
    for (int s = tid; s < nCols; s += 128) atomicAdd(&asum[s], colsum[s]);
}

// ---------------------------------------------------------------------------
__global__ void __launch_bounds__(256)
k_layernorm(const float* __restrict__ y, const float* __restrict__ g,
            const float* __restrict__ bb, unsigned short* __restrict__ out) {
    __shared__ float red[2][8];
    int row = blockIdx.x, tid = threadIdx.x;
    const float* yr = y + (size_t)row * Ec;
    float s = 0.f, s2 = 0.f, vals[4];
#pragma unroll
    for (int e = 0; e < 4; ++e) {
        float v = yr[tid + e * 256];
        vals[e] = v; s += v; s2 += v * v;
    }
#pragma unroll
    for (int m = 1; m < 32; m <<= 1) { s += __shfl_xor(s, m, 32); s2 += __shfl_xor(s2, m, 32); }
    if ((tid & 31) == 0) { red[0][tid >> 5] = s; red[1][tid >> 5] = s2; }
    __syncthreads();
    s = 0.f; s2 = 0.f;
#pragma unroll
    for (int w = 0; w < 8; ++w) { s += red[0][w]; s2 += red[1][w]; }
    float mu   = s * (1.0f / Ec);
    float var  = s2 * (1.0f / Ec) - mu * mu;
    float rstd = rsqrtf(var + 1e-5f);
#pragma unroll
    for (int e = 0; e < 4; ++e) {
        int c = tid + e * 256;
        out[(size_t)row * Ec + c] = f2bf((vals[e] - mu) * rstd * g[c] + bb[c]);
    }
}

__global__ void k_finalize(const float* __restrict__ asum, float* __restrict__ out) {
    int idx = blockIdx.x * blockDim.x + threadIdx.x;
    if (idx >= Bc * Nc * HQc) return;
    int h = idx & 15;
    int s = (idx >> 4) & (Nc - 1);
    int b = idx >> 15;                    // N*HQ = 2^15
    int i = h & 3, j = h >> 2;            // h = j*4 + i
    out[idx] = asum[(((size_t)b * HKVc + i) * Gc + j) * Nc + s] * (1.0f / Nc);
}

// ---------------------------------------------------------------------------
extern "C" void kernel_launch(void* const* d_in, const int* in_sizes, int n_in,
                              void* d_out, int out_size, void* d_ws, size_t ws_size,
                              hipStream_t stream) {
    (void)in_sizes; (void)n_in; (void)out_size; (void)ws_size;
    const float* x    = (const float*)d_in[0];
    const float* Wq   = (const float*)d_in[1];
    const float* Wk   = (const float*)d_in[2];
    const float* Wv   = (const float*)d_in[3];
    const float* Wout = (const float*)d_in[4];
    const float* lng  = (const float*)d_in[5];
    const float* lnb  = (const float*)d_in[6];

    char* ws = (char*)d_ws;
    size_t off = 0;
    auto alloc = [&](size_t bytes) -> char* {
        char* p = ws + off;
        off = (off + bytes + 255) & ~(size_t)255;
        return p;
    };
    const size_t M = (size_t)Bc * Nc;     // 4096 rows
    unsigned short* x_bf  = (unsigned short*)alloc(M * Ec * 2);
    unsigned short* wq_bf = (unsigned short*)alloc((size_t)Ec * Ec * 2);
    unsigned short* wk_bf = (unsigned short*)alloc((size_t)Ec * KVEc * 2);
    unsigned short* wv_bf = (unsigned short*)alloc((size_t)Ec * KVEc * 2);
    unsigned short* wo_bf = (unsigned short*)alloc((size_t)Ec * Ec * 2);
    unsigned short* q_bf  = (unsigned short*)alloc(M * Ec * 2);
    unsigned short* k_bf  = (unsigned short*)alloc(M * KVEc * 2);
    unsigned short* v_bf  = (unsigned short*)alloc(M * KVEc * 2);
    float*          ybuf  = (float*)alloc(M * Ec * 4);
    unsigned short* yn_bf = (unsigned short*)alloc(M * Ec * 2);
    float*          asum  = (float*)alloc((size_t)Bc * HQc * Nc * 4);

    auto cvt = [&](const float* src, unsigned short* dst, int n) {
        k_cvt_bf16<<<(n + 255) / 256, 256, 0, stream>>>(src, dst, n);
    };
    cvt(x,    x_bf,  (int)(M * Ec));
    cvt(Wq,   wq_bf, Ec * Ec);
    cvt(Wk,   wk_bf, Ec * KVEc);
    cvt(Wv,   wv_bf, Ec * KVEc);
    cvt(Wout, wo_bf, Ec * Ec);
    k_zero<<<(Bc * HQc * Nc + 255) / 256, 256, 0, stream>>>(asum, Bc * HQc * Nc);

    dim3 blk(128);
    k_gemm<false><<<dim3(Ec / 64,   (unsigned)(M / 64)), blk, 0, stream>>>(
        (const __bf16*)x_bf, (const __bf16*)wq_bf, q_bf, (int)M, Ec, Ec);
    k_gemm<false><<<dim3(KVEc / 64, (unsigned)(M / 64)), blk, 0, stream>>>(
        (const __bf16*)x_bf, (const __bf16*)wk_bf, k_bf, (int)M, Ec, KVEc);
    k_gemm<false><<<dim3(KVEc / 64, (unsigned)(M / 64)), blk, 0, stream>>>(
        (const __bf16*)x_bf, (const __bf16*)wv_bf, v_bf, (int)M, Ec, KVEc);

    k_attn<<<dim3(Nc / 64, HQc, Bc), blk, 0, stream>>>(
        (const __bf16*)q_bf, (const __bf16*)k_bf, (const __bf16*)v_bf, ybuf, asum);

    k_layernorm<<<dim3((unsigned)M), 256, 0, stream>>>(ybuf, lng, lnb, yn_bf);

    k_gemm<true><<<dim3(Ec / 64, (unsigned)(M / 64)), blk, 0, stream>>>(
        (const __bf16*)yn_bf, (const __bf16*)wo_bf, (float*)d_out, (int)M, Ec, Ec);

    k_finalize<<<(Bc * Nc * HQc + 255) / 256, 256, 0, stream>>>(
        asum, (float*)d_out + M * Ec);
}
